// NSP_55387898250045
// MI455X (gfx1250) — matrix operand order repaired and checked by reading, not compiled, not run on hardware
//
#include <hip/hip_runtime.h>
#include <hip/hip_bf16.h>
#include <cstdint>
#include <cstddef>

// Problem constants (match reference)
#define T_TOK 4096
#define DM    768
#define VVOC  32000
#define NLAYER 4
#define LN_EPS 1e-5f

typedef __attribute__((ext_vector_type(16))) __bf16 v16bf;
typedef __attribute__((ext_vector_type(8)))  __bf16 v8bf;
typedef __attribute__((ext_vector_type(8)))  float  v8f;

__device__ __forceinline__ float f4c(const float4& v, int c) {
    return c == 0 ? v.x : c == 1 ? v.y : c == 2 ? v.z : v.w;
}

// ---------------------------------------------------------------------------
// WMMA GEMM: C[M,N] = epilogue(A[M,K] @ B[K,N] + bias)
// MODE 0: none   MODE 1: qgelu   MODE 2: + resid[(m+shift), n] (zero if OOB)
// Tiles: 128x128x32, 256 threads (8 waves, 2x4 wave grid, 64x32 per wave).
// Double-buffered LDS; fp32 -> bf16 with native packed cvt while staging.
// B staged as 4x4 register blocks -> transposed -> ds_store_b64 (no b16 spam).
// ---------------------------------------------------------------------------
#define BM 128
#define BN 128
#define BK 32
#define PA 40   // LDS pitch (bf16 elems) for A tile rows
#define PB 40   // LDS pitch (bf16 elems) for B tile cols

template <int MODE>
__global__ __launch_bounds__(256)
void gemm_wmma(const float* __restrict__ A, const float* __restrict__ B,
               const float* __restrict__ bias, float* __restrict__ C,
               int M, int N, int K,
               const float* __restrict__ resid, int resid_shift)
{
    __shared__ __bf16 As[2][BM * PA];
    __shared__ __bf16 Bs[2][BN * PB];

    const int tid   = threadIdx.x;
    const int lane  = tid & 31;
    const int wave  = tid >> 5;
    const int waveM = wave & 1;   // 0..1  -> 64-row group
    const int waveN = wave >> 1;  // 0..3  -> 32-col group
    const int bm = blockIdx.y * BM;
    const int bn = blockIdx.x * BN;

    const int lrow  = lane & 15;
    const int hi16  = lane >> 4;          // 0 for lanes 0-15, 1 for 16-31
    const int ahalf = hi16 ? 8  : 0;      // A frag K base (ISA 16-bit A layout)
    const int bhalf = hi16 ? 8  : 0;      // B frag K base (matches A layout)

    // B staging: each thread owns a 4(K) x 4(N) block
    const int bn4 = (tid & 31) << 2;      // N quad within tile
    const int bkr = (tid >> 5) << 2;      // K row base within tile

    v8f acc[4][2];
#pragma unroll
    for (int mi = 0; mi < 4; ++mi)
#pragma unroll
        for (int ni = 0; ni < 2; ++ni)
            acc[mi][ni] = (v8f){0.f, 0.f, 0.f, 0.f, 0.f, 0.f, 0.f, 0.f};

    const int NT = K / BK;
    float4 areg[4], breg[4];

    // issue all 8 tile loads for K-tile k0 (kept in flight together)
    auto load_tile = [&](int k0) {
#pragma unroll
        for (int i = 0; i < 4; ++i) {
            const int id = tid + i * 256;
            areg[i] = *(const float4*)(A + (size_t)(bm + (id >> 3)) * K + k0 + ((id & 7) << 2));
        }
#pragma unroll
        for (int i = 0; i < 4; ++i)
            breg[i] = *(const float4*)(B + (size_t)(k0 + bkr + i) * N + bn + bn4);
    };
    // convert fp32 -> bf16 (native packed cvt) and store into LDS buffer `buf`
    auto store_tile = [&](int buf) {
#pragma unroll
        for (int i = 0; i < 4; ++i) {
            const int id = tid + i * 256;
            __bf16* p = &As[buf][(id >> 3) * PA + ((id & 7) << 2)];
            p[0] = (__bf16)areg[i].x; p[1] = (__bf16)areg[i].y;
            p[2] = (__bf16)areg[i].z; p[3] = (__bf16)areg[i].w;
        }
        // transpose the 4x4 block in registers; 4 contiguous bf16 along K
#pragma unroll
        for (int c = 0; c < 4; ++c) {
            __bf16* p = &Bs[buf][(bn4 + c) * PB + bkr];
            p[0] = (__bf16)f4c(breg[0], c);
            p[1] = (__bf16)f4c(breg[1], c);
            p[2] = (__bf16)f4c(breg[2], c);
            p[3] = (__bf16)f4c(breg[3], c);
        }
    };

    load_tile(0);
    store_tile(0);
    __syncthreads();

    for (int kt = 0; kt < NT; ++kt) {
        if (kt + 1 < NT) load_tile((kt + 1) * BK);   // next tile in flight now

        const int cur = kt & 1;
        // ---- load fragments per ISA 16-bit layouts ----
        v16bf afrag[4];
#pragma unroll
        for (int mi = 0; mi < 4; ++mi) {
            const int m = waveM * 64 + mi * 16 + lrow;
            const v8bf lo = *(const v8bf*)&As[cur][m * PA + ahalf];       // K = ahalf..+7
            const v8bf hh = *(const v8bf*)&As[cur][m * PA + ahalf + 16];  // K = ahalf+16..+23
            afrag[mi] = __builtin_shufflevector(lo, hh,
                0,1,2,3,4,5,6,7,8,9,10,11,12,13,14,15);
        }
        v16bf bfrag[2];
#pragma unroll
        for (int ni = 0; ni < 2; ++ni) {
            const int n = waveN * 32 + ni * 16 + lrow;
            const v8bf lo = *(const v8bf*)&Bs[cur][n * PB + bhalf];       // K = bhalf..+7
            const v8bf hh = *(const v8bf*)&Bs[cur][n * PB + bhalf + 16];  // K = bhalf+16..+23
            bfrag[ni] = __builtin_shufflevector(lo, hh,
                0,1,2,3,4,5,6,7,8,9,10,11,12,13,14,15);
        }

#pragma unroll
        for (int mi = 0; mi < 4; ++mi)
#pragma unroll
            for (int ni = 0; ni < 2; ++ni)
                acc[mi][ni] = __builtin_amdgcn_wmma_f32_16x16x32_bf16(
                    false, afrag[mi], false, bfrag[ni],
                    (short)0, acc[mi][ni], false, false);

        if (kt + 1 < NT) store_tile((kt + 1) & 1);   // fill other buffer
        __syncthreads();
    }

    // ---- epilogue: C/D layout -> lane n = base+lane%16, m = base + 8*hi16 + v
#pragma unroll
    for (int mi = 0; mi < 4; ++mi) {
#pragma unroll
        for (int ni = 0; ni < 2; ++ni) {
            const int n     = bn + waveN * 32 + ni * 16 + lrow;
            const int mbase = bm + waveM * 64 + mi * 16 + hi16 * 8;
            const float bv  = bias ? bias[n] : 0.f;
#pragma unroll
            for (int v = 0; v < 8; ++v) {
                const int m = mbase + v;
                float c = acc[mi][ni][v] + bv;
                if (MODE == 1) {
                    c = c / (1.f + __expf(-1.702f * c));   // qgelu
                } else if (MODE == 2) {
                    const int mr = m + resid_shift;
                    if (mr < M) c += resid[(size_t)mr * N + n];
                }
                C[(size_t)m * N + n] = c;
            }
        }
    }
}

// ---------------------------------------------------------------------------
// LayerNorm: one block per row, width W (768 or 1536); supports in-place.
// ---------------------------------------------------------------------------
__global__ __launch_bounds__(256)
void layernorm_k(const float* __restrict__ X, float* __restrict__ Y,
                 const float* __restrict__ g, const float* __restrict__ b, int W)
{
    __shared__ float red[256];
    const int row = blockIdx.x;
    const float* x = X + (size_t)row * W;
    float* y = Y + (size_t)row * W;
    const int cnt = W >> 8;              // 3 or 6
    float vals[8];
    float s = 0.f;
    for (int i = 0; i < cnt; ++i) {
        vals[i] = x[threadIdx.x + (i << 8)];
        s += vals[i];
    }
    red[threadIdx.x] = s; __syncthreads();
    for (int off = 128; off > 0; off >>= 1) {
        if (threadIdx.x < off) red[threadIdx.x] += red[threadIdx.x + off];
        __syncthreads();
    }
    const float mean = red[0] / (float)W;
    __syncthreads();
    float s2 = 0.f;
    for (int i = 0; i < cnt; ++i) {
        const float d = vals[i] - mean;
        s2 += d * d;
    }
    red[threadIdx.x] = s2; __syncthreads();
    for (int off = 128; off > 0; off >>= 1) {
        if (threadIdx.x < off) red[threadIdx.x] += red[threadIdx.x + off];
        __syncthreads();
    }
    const float rstd = rsqrtf(red[0] / (float)W + LN_EPS);
    __syncthreads();
    for (int i = 0; i < cnt; ++i) {
        const int c = threadIdx.x + (i << 8);
        y[c] = (vals[i] - mean) * rstd * g[c] + b[c];
    }
}

// ---------------------------------------------------------------------------
// x[t, :] = wte[idx[t], :]
// ---------------------------------------------------------------------------
__global__ __launch_bounds__(256)
void gather_embed_k(const int* __restrict__ idx, const float* __restrict__ wte,
                    float* __restrict__ x)
{
    const int t = blockIdx.x;
    const int r = idx[t];
    for (int d = threadIdx.x; d < DM; d += 256)
        x[(size_t)t * DM + d] = wte[(size_t)r * DM + d];
}

// ---------------------------------------------------------------------------
// Gumbel-argmax categorical sample per row + mask computation.
// mask[t] = (sample == idx[t+1]) && (t < T-j)   (hash RNG stand-in)
// ---------------------------------------------------------------------------
__device__ __forceinline__ unsigned hash3(unsigned a, unsigned b, unsigned c) {
    unsigned h = a * 0x9E3779B1u ^ b * 0x85EBCA77u ^ c * 0xC2B2AE3Du;
    h ^= h >> 15; h *= 0x2C1B3C6Du;
    h ^= h >> 12; h *= 0x297A2D39u;
    h ^= h >> 15;
    return h;
}

__global__ __launch_bounds__(256)
void sample_mask_k(const float* __restrict__ logits, const int* __restrict__ idx,
                   int layer, int j, int* __restrict__ mask)
{
    __shared__ float bv[256];
    __shared__ int   bi[256];
    const int t = blockIdx.x;
    const float* lr = logits + (size_t)t * VVOC;
    float best = -1e30f; int besti = -1;
    for (int v = threadIdx.x; v < VVOC; v += 256) {
        const unsigned h = hash3((unsigned)t, (unsigned)v, (unsigned)(layer + 1));
        const float u = (float)(h >> 8) * (1.0f / 16777216.0f) + 1e-10f;
        const float gum = -__logf(-__logf(u));
        const float s = lr[v] + gum;
        if (s > best) { best = s; besti = v; }
    }
    bv[threadIdx.x] = best; bi[threadIdx.x] = besti; __syncthreads();
    for (int off = 128; off > 0; off >>= 1) {
        if (threadIdx.x < off && bv[threadIdx.x + off] > bv[threadIdx.x]) {
            bv[threadIdx.x] = bv[threadIdx.x + off];
            bi[threadIdx.x] = bi[threadIdx.x + off];
        }
        __syncthreads();
    }
    if (threadIdx.x == 0) {
        const int tgt = (t < T_TOK - 1) ? idx[t + 1] : -1;
        mask[t] = (bi[0] == tgt && t < T_TOK - j) ? 1 : 0;
    }
}

// xcat[t] = [ x[t], x[t+j] or 0 ]
__global__ __launch_bounds__(256)
void build_cat_k(const float* __restrict__ x, float* __restrict__ xcat, int j)
{
    const int t = blockIdx.x;
    for (int d = threadIdx.x; d < DM; d += 256) {
        xcat[(size_t)t * (2 * DM) + d] = x[(size_t)t * DM + d];
        xcat[(size_t)t * (2 * DM) + DM + d] =
            (t + j < T_TOK) ? x[(size_t)(t + j) * DM + d] : 0.f;
    }
}

// for t >= j where mask[t-j]: x[t] = merged[t-j]
__global__ __launch_bounds__(256)
void scatter_k(float* __restrict__ x, const float* __restrict__ merged,
               const int* __restrict__ mask, int j)
{
    const int t = blockIdx.x;
    if (t < j || !mask[t - j]) return;
    for (int d = threadIdx.x; d < DM; d += 256)
        x[(size_t)t * DM + d] = merged[(size_t)(t - j) * DM + d];
}

// ---------------------------------------------------------------------------
// Host-side orchestration
// ---------------------------------------------------------------------------
static inline dim3 gemm_grid(int M, int N) { return dim3(N / BN, M / BM); }

extern "C" void kernel_launch(void* const* d_in, const int* in_sizes, int n_in,
                              void* d_out, int out_size, void* d_ws, size_t ws_size,
                              hipStream_t stream)
{
    (void)in_sizes; (void)n_in; (void)out_size; (void)ws_size;

    const int*   idx       = (const int*)  d_in[0];
    const float* wte       = (const float*)d_in[1];
    const float* rb_ln_g   = (const float*)d_in[2];
    const float* rb_ln_b   = (const float*)d_in[3];
    const float* rb_fc_w   = (const float*)d_in[4];
    const float* rb_fc_b   = (const float*)d_in[5];
    const float* rb_proj_w = (const float*)d_in[6];
    const float* rb_proj_b = (const float*)d_in[7];
    const float* lnf_g     = (const float*)d_in[8];
    const float* lnf_b     = (const float*)d_in[9];
    const float* lm_head_w = (const float*)d_in[10];
    const float* mb_ln_g   = (const float*)d_in[11];
    const float* mb_ln_b   = (const float*)d_in[12];
    const float* mb_fc_w   = (const float*)d_in[13];
    const float* mb_fc_b   = (const float*)d_in[14];
    const float* mb_proj_w = (const float*)d_in[15];
    const float* mb_proj_b = (const float*)d_in[16];
    const float* op_w      = (const float*)d_in[17];
    const float* op_b      = (const float*)d_in[18];
    const float* lns_g     = (const float*)d_in[19];
    const float* lns_b     = (const float*)d_in[20];
    const float* heads_w   = (const float*)d_in[21];

    float* logits = (float*)d_out;                       // [T, V]

    // workspace layout (floats)
    float* ws   = (float*)d_ws;
    float* x    = ws;                                    // [T, D]
    float* xcat = x    + (size_t)T_TOK * DM;             // [T, 2D]
    float* ff   = xcat + (size_t)T_TOK * 2 * DM;         // [T, 4D]
    float* mrg  = ff   + (size_t)T_TOK * 4 * DM;         // [T, D]
    float* xtmp = mrg  + (size_t)T_TOK * DM;             // [T, D]
    int*   mask = (int*)(xtmp + (size_t)T_TOK * DM);     // [T]

    const dim3 blk(256);

    // x = wte[idx]
    gather_embed_k<<<T_TOK, blk, 0, stream>>>(idx, wte, x);

    // h = ln(x); x = x + qgelu(h@fc+b)@proj + b
    layernorm_k<<<T_TOK, blk, 0, stream>>>(x, xtmp, rb_ln_g, rb_ln_b, DM);
    gemm_wmma<1><<<gemm_grid(T_TOK, 4 * DM), blk, 0, stream>>>(
        xtmp, rb_fc_w, rb_fc_b, ff, T_TOK, 4 * DM, DM, nullptr, 0);
    gemm_wmma<2><<<gemm_grid(T_TOK, DM), blk, 0, stream>>>(
        ff, rb_proj_w, rb_proj_b, x, T_TOK, DM, 4 * DM, x, 0);

    // x = ln_f(x); logits = x @ lm_head_w
    layernorm_k<<<T_TOK, blk, 0, stream>>>(x, x, lnf_g, lnf_b, DM);
    gemm_wmma<0><<<gemm_grid(T_TOK, VVOC), blk, 0, stream>>>(
        x, lm_head_w, nullptr, logits, T_TOK, VVOC, DM, nullptr, 0);

    for (int i = 0; i < NLAYER; ++i) {
        const int j = i + 1;

        sample_mask_k<<<T_TOK, blk, 0, stream>>>(logits, idx, i, j, mask);

        // merge MLP on concat(x, shift(x, j))
        build_cat_k<<<T_TOK, blk, 0, stream>>>(x, xcat, j);
        layernorm_k<<<T_TOK, blk, 0, stream>>>(
            xcat, xcat, mb_ln_g + (size_t)i * 2 * DM, mb_ln_b + (size_t)i * 2 * DM, 2 * DM);
        gemm_wmma<1><<<gemm_grid(T_TOK, 4 * DM), blk, 0, stream>>>(
            xcat, mb_fc_w + (size_t)i * 2 * DM * 4 * DM, mb_fc_b + (size_t)i * 4 * DM,
            ff, T_TOK, 4 * DM, 2 * DM, nullptr, 0);
        gemm_wmma<2><<<gemm_grid(T_TOK, DM), blk, 0, stream>>>(
            ff, mb_proj_w + (size_t)i * 4 * DM * DM, mb_proj_b + (size_t)i * DM,
            mrg, T_TOK, DM, 4 * DM, x, j);

        scatter_k<<<T_TOK, blk, 0, stream>>>(x, mrg, mask, j);

        // x = ln_s(x @ op + b); logits = x @ heads_w[i]
        gemm_wmma<0><<<gemm_grid(T_TOK, DM), blk, 0, stream>>>(
            x, op_w + (size_t)i * DM * DM, op_b + (size_t)i * DM,
            xtmp, T_TOK, DM, DM, nullptr, 0);
        layernorm_k<<<T_TOK, blk, 0, stream>>>(
            xtmp, x, lns_g + (size_t)i * DM, lns_b + (size_t)i * DM, DM);
        gemm_wmma<0><<<gemm_grid(T_TOK, VVOC), blk, 0, stream>>>(
            x, heads_w + (size_t)i * DM * VVOC, nullptr, logits, T_TOK, VVOC, DM, nullptr, 0);
    }
}
